// SAM_77653008712339
// MI455X (gfx1250) — matrix operand order repaired and checked by reading, not compiled, not run on hardware
//
#include <hip/hip_runtime.h>
#include <hip/hip_bf16.h>
#include <math.h>

typedef __bf16 bf16;
typedef __attribute__((ext_vector_type(16))) __bf16 v16bf;
typedef __attribute__((ext_vector_type(8)))  __bf16 v8bf;
typedef __attribute__((ext_vector_type(8)))  float  v8f;

#define B_ 4
#define C_ 256
#define D_ 64
#define L_ 4096
#define KT 64
#define LOG2E 1.4426950408889634f

static __device__ __forceinline__ bf16 f2bf(float f) { return (bf16)f; }

static __device__ __forceinline__ v8bf ld8(const bf16* p) { return *(const v8bf*)p; }

static __device__ __forceinline__ v16bf mk16(v8bf lo, v8bf hi) {
  v16bf r;
#pragma unroll
  for (int i = 0; i < 8; ++i) { r[i] = lo[i]; r[i + 8] = hi[i]; }
  return r;
}

static __device__ __forceinline__ v8f wmma_bf16(v16bf a, v16bf b, v8f c) {
  return __builtin_amdgcn_wmma_f32_16x16x32_bf16(false, a, false, b,
                                                 (short)0, c, false, false);
}

// ---- CDNA5 async global->LDS copy (ASYNCcnt-tracked), 16 bytes per lane ----
static __device__ __forceinline__ unsigned lds_off(const void* p) {
  // shared-aperture generic pointers keep the LDS byte offset in the low 32 bits
  return (unsigned)(uintptr_t)p;
}
static __device__ __forceinline__ void async_cp16(unsigned lds, const void* g) {
  asm volatile("global_load_async_to_lds_b128 %0, %1, off"
               :: "v"(lds), "v"(g) : "memory");
}
static __device__ __forceinline__ void wait_async0() {
  asm volatile("s_wait_asynccnt 0" ::: "memory");
}
static __device__ __forceinline__ void wait_ds0() {
  asm volatile("s_wait_dscnt 0" ::: "memory");
}

// -------------------------------------------------------------------------
// Kernel 1: Q/K/V projections. Q,K stored [pos][d]; V stored transposed
// [d][pos] so the attention kernel can async-copy both operands untouched.
// Q additionally pre-scaled by log2(e) so softmax can use raw v_exp_f32.
// -------------------------------------------------------------------------
__global__ __launch_bounds__(128)
void qkv_proj_kernel(const float* __restrict__ mask, const float* __restrict__ edge,
                     const float* __restrict__ WQ, const float* __restrict__ bQ,
                     const float* __restrict__ WK, const float* __restrict__ bK,
                     const float* __restrict__ WV, const float* __restrict__ bV,
                     bf16* __restrict__ Qw, bf16* __restrict__ Kw, bf16* __restrict__ Vt) {
  const int tilesPerSel = L_ / 64;  // 64
  int bid  = blockIdx.x;
  int n    = bid / (3 * tilesPerSel);
  int r    = bid % (3 * tilesPerSel);
  int sel  = r / tilesPerSel;
  int tile = r % tilesPerSel;

  int tid  = threadIdx.x;
  int wave = tid >> 5;
  int lane = tid & 31;
  int lc   = lane & 15;
  int half = lane >> 4;
  int pos0 = tile * 64 + wave * 16;

  const float* X  = (sel == 0 ? edge : mask) + (size_t)n * C_ * L_;
  const float* Wm = (sel == 0 ? WQ : (sel == 1 ? WK : WV));
  const float* bv = (sel == 0 ? bQ : (sel == 1 ? bK : bV));
  bf16* Ow = (sel == 0 ? Qw : (sel == 1 ? Kw : Vt)) + (size_t)n * L_ * D_;
  float scale = (sel == 0) ? LOG2E : 1.0f;

  // B fragments B[k=c][col=pos] for all 8 K-chunks
  v16bf bfr[8];
#pragma unroll
  for (int ch = 0; ch < 8; ++ch) {
    const float* xp = X + (size_t)(ch * 32 + half * 16) * L_ + pos0 + lc;
    v16bf t;
#pragma unroll
    for (int e = 0; e < 16; ++e) t[e] = f2bf(xp[(size_t)e * L_]);
    bfr[ch] = t;
  }

#pragma unroll
  for (int it = 0; it < 4; ++it) {
    v8f acc = {};
#pragma unroll
    for (int ch = 0; ch < 8; ++ch) {
      const float* wr = Wm + (size_t)(it * 16 + lc) * C_ + ch * 32;
      v16bf a;
#pragma unroll
      for (int e = 0; e < 8; ++e) {
        a[e]     = f2bf(wr[half * 16 + e]);
        a[e + 8] = f2bf(wr[half * 16 + 8 + e]);
      }
      acc = wmma_bf16(a, bfr[ch], acc);
    }
#pragma unroll
    for (int j = 0; j < 8; ++j) {
      int i   = it * 16 + j + 8 * half;
      float y = fmaxf(acc[j] + bv[i], 0.0f) * scale;
      if (sel == 2)
        Ow[(size_t)i * L_ + pos0 + lc] = f2bf(y);       // V^T [d][pos]
      else
        Ow[(size_t)(pos0 + lc) * D_ + i] = f2bf(y);     // Q,K [pos][d]
    }
  }
}

// -------------------------------------------------------------------------
// Kernel 2: flash attention, double-buffered async K/V staging.
// 8 waves/block, 16 query rows/wave, 64-key tiles, online softmax with
// WMMA row-sum (all-ones B) and exp2-domain max tracking.
// -------------------------------------------------------------------------
__global__ __launch_bounds__(256)
void attn_kernel(const bf16* __restrict__ Qw, const bf16* __restrict__ Kw,
                 const bf16* __restrict__ Vt, bf16* __restrict__ Yw) {
  __shared__ bf16 Ks[2][KT * D_];    // [buf][key][d]   8KB each
  __shared__ bf16 VTs[2][D_ * KT];   // [buf][d][key]   8KB each
  __shared__ bf16 Ps[8][16 * KT];    // per-wave P repack [q][k] 2KB each

  const int qBlocks = L_ / 128;      // 32
  int bid = blockIdx.x;
  int n   = bid / qBlocks;
  int qb  = bid % qBlocks;

  int tid  = threadIdx.x;
  int wave = tid >> 5;
  int lane = tid & 31;
  int lc   = lane & 15;
  int half = lane >> 4;
  int q0   = qb * 128 + wave * 16;

  const bf16* Qb = Qw + (size_t)n * L_ * D_;
  const bf16* Kb = Kw + (size_t)n * L_ * D_;
  const bf16* Vb = Vt + (size_t)n * L_ * D_;   // [d][L]
  bf16*       Yb = Yw + (size_t)n * L_ * D_;

  // issue one 64-key tile (K: 8KB contiguous, V^T: 64 rows of 128B)
  auto stage = [&](int buf, int kt) {
    const bf16* kg = Kb + (size_t)kt * D_;
    unsigned kl = lds_off(&Ks[buf][0]);
    async_cp16(kl + (unsigned)tid * 16u,         kg + (size_t)tid * 8);
    async_cp16(kl + (unsigned)(tid + 256) * 16u, kg + (size_t)(tid + 256) * 8);
    unsigned vl = lds_off(&VTs[buf][0]);
    int c0 = tid, c1 = tid + 256;
    async_cp16(vl + (unsigned)c0 * 16u, Vb + (size_t)(c0 >> 3) * L_ + kt + (c0 & 7) * 8);
    async_cp16(vl + (unsigned)c1 * 16u, Vb + (size_t)(c1 >> 3) * L_ + kt + (c1 & 7) * 8);
  };

  // Q A-fragments (2 K-chunks of 32 along d)
  v16bf qa[2];
#pragma unroll
  for (int ch = 0; ch < 2; ++ch) {
    const bf16* qp = Qb + (size_t)(q0 + lc) * D_ + ch * 32;
    qa[ch] = mk16(ld8(qp + half * 8), ld8(qp + 16 + half * 8));
  }

  // all-ones B fragment for WMMA row-sums
  v16bf ones;
#pragma unroll
  for (int e = 0; e < 16; ++e) ones[e] = f2bf(1.0f);

  v8f acc[4] = {};
  v8f accl = {};
  float mrow[8];
#pragma unroll
  for (int j = 0; j < 8; ++j) mrow[j] = -INFINITY;

  stage(0, 0);

  for (int kt = 0; kt < L_; kt += KT) {
    int cur = (kt / KT) & 1;
    wait_async0();     // my async copies for buffer `cur` have landed
    wait_ds0();        // my LDS reads of the other buffer have returned
    __syncthreads();   // everyone's tile is in, everyone's reads are done
    if (kt + KT < L_) stage(cur ^ 1, kt + KT);

    const bf16* ks = &Ks[cur][0];
    const bf16* vs = &VTs[cur][0];

    // S = Q K^T : four 16x16 key subtiles, K-dim 64 (2 chunks)
    v8f s[4];
#pragma unroll
    for (int st = 0; st < 4; ++st) {
      v8f sa = {};
#pragma unroll
      for (int ch = 0; ch < 2; ++ch) {
        const bf16* kp = ks + (size_t)(st * 16 + lc) * D_ + ch * 32;
        sa = wmma_bf16(qa[ch], mk16(ld8(kp + half * 8), ld8(kp + 16 + half * 8)), sa);
      }
      s[st] = sa;
    }

    // online softmax (exp2 domain; Q pre-scaled by log2e)
    bf16* pp = &Ps[wave][0];
#pragma unroll
    for (int j = 0; j < 8; ++j) {
      float mx = fmaxf(fmaxf(s[0][j], s[1][j]), fmaxf(s[2][j], s[3][j]));
#pragma unroll
      for (int off = 8; off >= 1; off >>= 1)
        mx = fmaxf(mx, __shfl_xor(mx, off, 16));
      float mnew = fmaxf(mrow[j], mx);
      float sc   = __builtin_amdgcn_exp2f(mrow[j] - mnew);
      mrow[j]    = mnew;
      int rr = j + 8 * half;
#pragma unroll
      for (int st = 0; st < 4; ++st) {
        float p = __builtin_amdgcn_exp2f(s[st][j] - mnew);
        pp[rr * KT + st * 16 + lc] = f2bf(p);
      }
      acc[0][j] *= sc; acc[1][j] *= sc; acc[2][j] *= sc; acc[3][j] *= sc;
      accl[j] *= sc;
    }
    wait_ds0();  // P repack store->load ordering

    // P A-fragments (16 q x 64 keys -> 2 chunks of 32)
    const bf16* prow = pp + (size_t)lc * KT;
    v16bf pa0 = mk16(ld8(prow + half * 8),      ld8(prow + 16 + half * 8));
    v16bf pa1 = mk16(ld8(prow + 32 + half * 8), ld8(prow + 48 + half * 8));

    // acc += P * V ; accl += P * ones (row sums)
#pragma unroll
    for (int nt = 0; nt < 4; ++nt) {
      const bf16* vr = vs + (size_t)(nt * 16 + lc) * KT;
      acc[nt] = wmma_bf16(pa0, mk16(ld8(vr + half * 8), ld8(vr + 16 + half * 8)), acc[nt]);
      acc[nt] = wmma_bf16(pa1, mk16(ld8(vr + 32 + half * 8), ld8(vr + 48 + half * 8)), acc[nt]);
    }
    accl = wmma_bf16(pa0, ones, accl);
    accl = wmma_bf16(pa1, ones, accl);
  }

  // finalize: divide by row sum, store Y[pos][d] bf16
#pragma unroll
  for (int j = 0; j < 8; ++j) {
    float inv = 1.0f / accl[j];
    size_t row = (size_t)(q0 + j + 8 * half) * D_;
#pragma unroll
    for (int nt = 0; nt < 4; ++nt)
      Yb[row + nt * 16 + lc] = f2bf(acc[nt][j] * inv);
  }
}

// -------------------------------------------------------------------------
// Kernel 3: out = relu(WO y + bO) + mask_feats.  M = 256 co, K = 64, N = pos.
// -------------------------------------------------------------------------
__global__ __launch_bounds__(256)
void outproj_kernel(const bf16* __restrict__ Yw, const float* __restrict__ WO,
                    const float* __restrict__ bO, const float* __restrict__ mask,
                    float* __restrict__ out) {
  const int ptiles = L_ / 16;  // 256
  int bid  = blockIdx.x;
  int n    = bid / ptiles;
  int pos0 = (bid % ptiles) * 16;

  int tid  = threadIdx.x;
  int wave = tid >> 5;
  int lane = tid & 31;
  int lc   = lane & 15;
  int half = lane >> 4;

  const bf16* Yb = Yw + (size_t)n * L_ * D_;

  v16bf bfr[2];
#pragma unroll
  for (int ch = 0; ch < 2; ++ch) {
    const bf16* yp = Yb + (size_t)(pos0 + lc) * D_ + ch * 32 + half * 16;
    bfr[ch] = mk16(ld8(yp), ld8(yp + 8));
  }

#pragma unroll
  for (int ct = 0; ct < 2; ++ct) {
    int cot = wave * 2 + ct;
    v8f acc = {};
#pragma unroll
    for (int ch = 0; ch < 2; ++ch) {
      const float* wr = WO + (size_t)(cot * 16 + lc) * D_ + ch * 32;
      v16bf a;
#pragma unroll
      for (int e = 0; e < 8; ++e) {
        a[e]     = f2bf(wr[half * 16 + e]);
        a[e + 8] = f2bf(wr[half * 16 + 8 + e]);
      }
      acc = wmma_bf16(a, bfr[ch], acc);
    }
#pragma unroll
    for (int j = 0; j < 8; ++j) {
      int co = cot * 16 + j + 8 * half;
      size_t idx = ((size_t)n * C_ + co) * L_ + pos0 + lc;
      out[idx] = fmaxf(acc[j] + bO[co], 0.0f) + mask[idx];
    }
  }
}

extern "C" void kernel_launch(void* const* d_in, const int* in_sizes, int n_in,
                              void* d_out, int out_size, void* d_ws, size_t ws_size,
                              hipStream_t stream) {
  const float* mask = (const float*)d_in[0];
  const float* edge = (const float*)d_in[1];
  const float* WQ   = (const float*)d_in[2];
  const float* bQ   = (const float*)d_in[3];
  const float* WK   = (const float*)d_in[4];
  const float* bK   = (const float*)d_in[5];
  const float* WV   = (const float*)d_in[6];
  const float* bV   = (const float*)d_in[7];
  const float* WO   = (const float*)d_in[8];
  const float* bO   = (const float*)d_in[9];
  float* out = (float*)d_out;

  const size_t mat = (size_t)B_ * L_ * D_;  // 1M elems per matrix (bf16)
  bf16* Qw = (bf16*)d_ws;
  bf16* Kw = Qw + mat;
  bf16* Vt = Kw + mat;
  bf16* Yw = Vt + mat;  // 8 MB workspace total

  qkv_proj_kernel<<<B_ * 3 * (L_ / 64), 128, 0, stream>>>(
      mask, edge, WQ, bQ, WK, bK, WV, bV, Qw, Kw, Vt);
  attn_kernel<<<B_ * (L_ / 128), 256, 0, stream>>>(Qw, Kw, Vt, Yw);
  outproj_kernel<<<B_ * (L_ / 16), 256, 0, stream>>>(Yw, WO, bO, mask, out);
}
